// GraphSAGE_26474178413285
// MI455X (gfx1250) — matrix operand order repaired and checked by reading, not compiled, not run on hardware
//
#include <hip/hip_runtime.h>

typedef __attribute__((ext_vector_type(16))) __bf16        v16bf;
typedef __attribute__((ext_vector_type(8)))  float         v8f;
typedef __attribute__((ext_vector_type(4)))  float         f4;
typedef __attribute__((ext_vector_type(4)))  unsigned int  u4;
typedef __attribute__((ext_vector_type(2)))  unsigned int  u2;

__device__ __forceinline__ unsigned short f2bfbits(float f) {
  unsigned int u = __builtin_bit_cast(unsigned int, f);
  u += 0x7fffu + ((u >> 16) & 1u);           // round-to-nearest-even
  return (unsigned short)(u >> 16);
}
__device__ __forceinline__ float bfbits2f(unsigned int b) {
  return __builtin_bit_cast(float, b << 16);
}

union AFragU { v16bf v; unsigned short u[16]; };
union BFragU { v16bf v; u4 q[2]; };

// A-fragment (16x32 bf16) from an f32 row with per-row scale (mean aggregation).
// Lane (half = lane>>4, mr = lane&15) holds row m0+mr,
// K = {k0+8*half .. +7} then {k0+16+8*half .. +7} (two contiguous 8-float runs).
__device__ __forceinline__ v16bf load_a_f32(const float* __restrict__ p, float s) {
  f4 a0 = ((const f4*)p)[0];
  f4 a1 = ((const f4*)p)[1];
  f4 a2 = ((const f4*)(p + 16))[0];
  f4 a3 = ((const f4*)(p + 16))[1];
  float t[16] = {a0.x, a0.y, a0.z, a0.w, a1.x, a1.y, a1.z, a1.w,
                 a2.x, a2.y, a2.z, a2.w, a3.x, a3.y, a3.z, a3.w};
  AFragU f;
#pragma unroll
  for (int i = 0; i < 16; ++i) f.u[i] = f2bfbits(t[i] * s);
  return f.v;
}

// A-fragment directly from pre-converted bf16 storage: two b128 loads, no VALU.
__device__ __forceinline__ v16bf load_a_bf(const unsigned short* __restrict__ p) {
  BFragU f;
  f.q[0] = ((const u4*)p)[0];        // K run 1: 8 contiguous bf16
  f.q[1] = *((const u4*)(p + 16));   // K run 2: 8 contiguous bf16
  return f.v;
}

// out[m,n] = sum_k (agg[m,k]/max(deg,1))*Wl[n,k] + bias[n] + sum_k feat[m,k]*Wr[n,k]
// One wave32 per *32-row* tile (two 16-row WMMA tiles): every B fragment is
// reused by 2 WMMAs, halving weight traffic per output vs one tile per wave.
template <int BFOUT, int RELU>
__global__ __launch_bounds__(128) void sage_gemm(
    const float* __restrict__ agg, const float* __restrict__ deg,
    const unsigned short* __restrict__ featbf,
    const unsigned short* __restrict__ Wl, const unsigned short* __restrict__ Wr,
    const float* __restrict__ bias,
    float* __restrict__ outF, unsigned short* __restrict__ outB,
    int N)
{
  const int wave = threadIdx.x >> 5;
  const int lane = threadIdx.x & 31;
  const int m0   = (blockIdx.x * 4 + wave) * 32;
  if (m0 >= N) return;
  const int half = lane >> 4;
  const int mr   = lane & 15;

  int r0 = m0 + mr;       if (r0 >= N) r0 = N - 1;   // clamp loads; guarded stores
  int r1 = m0 + 16 + mr;  if (r1 >= N) r1 = N - 1;
  const float rs0 = 1.0f / fmaxf(deg[r0], 1.0f);
  const float rs1 = 1.0f / fmaxf(deg[r1], 1.0f);

  const float*          aggR0 = agg    + (size_t)r0 * 128;
  const float*          aggR1 = agg    + (size_t)r1 * 128;
  const unsigned short* fR0   = featbf + (size_t)r0 * 128;
  const unsigned short* fR1   = featbf + (size_t)r1 * 128;

  v8f accA[8], accB[8];
  const v8f z8 = {0.f, 0.f, 0.f, 0.f, 0.f, 0.f, 0.f, 0.f};
#pragma unroll
  for (int nt = 0; nt < 8; ++nt) { accA[nt] = z8; accB[nt] = z8; }

#pragma unroll
  for (int kc = 0; kc < 128; kc += 32) {
    const int aOff = kc + 8 * half;
    const v16bf a1t0 = load_a_f32(aggR0 + aOff, rs0);  // mean-agg, rows m0..m0+15
    const v16bf a1t1 = load_a_f32(aggR1 + aOff, rs1);  // mean-agg, rows m0+16..m0+31
    const v16bf a2t0 = load_a_bf(fR0 + aOff);          // root feats (pre-bf16)
    const v16bf a2t1 = load_a_bf(fR1 + aOff);
    const int bOff = kc + 8 * half;
#pragma unroll
    for (int nt = 0; nt < 8; ++nt) {
      const int wrow = nt * 16 + mr;  // B[k,n] = W[n,k] -> contiguous W row segment
      BFragU b1, b2;
      const u4* p1 = (const u4*)(Wl + wrow * 128 + bOff);
      const u4* p2 = (const u4*)(Wr + wrow * 128 + bOff);
      b1.q[0] = p1[0]; b1.q[1] = p1[2];
      b2.q[0] = p2[0]; b2.q[1] = p2[2];
      accA[nt] = __builtin_amdgcn_wmma_f32_16x16x32_bf16(
          false, a1t0, false, b1.v, (short)0, accA[nt], false, false);
      accB[nt] = __builtin_amdgcn_wmma_f32_16x16x32_bf16(
          false, a1t1, false, b1.v, (short)0, accB[nt], false, false);
      accA[nt] = __builtin_amdgcn_wmma_f32_16x16x32_bf16(
          false, a2t0, false, b2.v, (short)0, accA[nt], false, false);
      accB[nt] = __builtin_amdgcn_wmma_f32_16x16x32_bf16(
          false, a2t1, false, b2.v, (short)0, accB[nt], false, false);
    }
  }

  // D layout: VGPR v -> row m0 + v + 8*half (+16 for tile B), col = nt*16 + mr
#pragma unroll
  for (int nt = 0; nt < 8; ++nt) {
    const int col = nt * 16 + mr;
    const float bv = bias[col];
#pragma unroll
    for (int v = 0; v < 8; ++v) {
      const int row0 = m0 + v + 8 * half;
      const int row1 = row0 + 16;
      if (row0 < N) {
        float val = accA[nt][v] + bv;
        if (RELU) val = fmaxf(val, 0.0f);
        if (BFOUT) outB[(unsigned)row0 * 128u + (unsigned)col] = f2bfbits(val);
        else       outF[(unsigned)row0 * 128u + (unsigned)col] = val;
      }
      if (row1 < N) {
        float val = accB[nt][v] + bv;
        if (RELU) val = fmaxf(val, 0.0f);
        if (BFOUT) outB[(unsigned)row1 * 128u + (unsigned)col] = f2bfbits(val);
        else       outF[(unsigned)row1 * 128u + (unsigned)col] = val;
      }
    }
  }
}

// One wave32 per edge: each lane gathers 4 bf16 channels (8B load), expands to
// f32, and atomically accumulates into the f32 aggregation buffer (L2-resident).
__global__ __launch_bounds__(256) void sage_scatter_bf(
    const unsigned short* __restrict__ featbf, const long long* __restrict__ ei,
    long long E, float* __restrict__ agg, float* __restrict__ deg, int withDeg)
{
  const long long wid = (long long)blockIdx.x * 8 + (threadIdx.x >> 5);
  if (wid >= E) return;
  const int lane = threadIdx.x & 31;
  const long long s = ei[wid];       // edge_index[0][e]
  const long long d = ei[E + wid];   // edge_index[1][e]
  const u2 w = ((const u2*)(featbf + s * 128))[lane];
  const float f0 = bfbits2f(w.x & 0xffffu);
  const float f1 = bfbits2f(w.x >> 16);
  const float f2 = bfbits2f(w.y & 0xffffu);
  const float f3 = bfbits2f(w.y >> 16);
  float* ap = agg + d * 128 + (long long)lane * 4;
  __hip_atomic_fetch_add(ap + 0, f0, __ATOMIC_RELAXED, __HIP_MEMORY_SCOPE_AGENT);
  __hip_atomic_fetch_add(ap + 1, f1, __ATOMIC_RELAXED, __HIP_MEMORY_SCOPE_AGENT);
  __hip_atomic_fetch_add(ap + 2, f2, __ATOMIC_RELAXED, __HIP_MEMORY_SCOPE_AGENT);
  __hip_atomic_fetch_add(ap + 3, f3, __ATOMIC_RELAXED, __HIP_MEMORY_SCOPE_AGENT);
  if (withDeg && lane == 0)
    __hip_atomic_fetch_add(deg + d, 1.0f, __ATOMIC_RELAXED, __HIP_MEMORY_SCOPE_AGENT);
}

__global__ void zero_f32(float* __restrict__ p, long long n) {
  long long i = (long long)blockIdx.x * blockDim.x + threadIdx.x;
  const long long stride = (long long)gridDim.x * blockDim.x;
  for (; i < n; i += stride) p[i] = 0.0f;
}

// Vectorized f32 -> packed bf16 conversion (4 elements / thread).
__global__ void cvt_bf16x4(const float* __restrict__ s, unsigned short* __restrict__ d,
                           long long n4) {
  const long long i = (long long)blockIdx.x * blockDim.x + threadIdx.x;
  if (i < n4) {
    const f4 v = ((const f4*)s)[i];
    u2 r;
    r.x = (unsigned)f2bfbits(v.x) | ((unsigned)f2bfbits(v.y) << 16);
    r.y = (unsigned)f2bfbits(v.z) | ((unsigned)f2bfbits(v.w) << 16);
    ((u2*)d)[i] = r;
  }
}

extern "C" void kernel_launch(void* const* d_in, const int* in_sizes, int n_in,
                              void* d_out, int out_size, void* d_ws, size_t ws_size,
                              hipStream_t stream) {
  const float*     x   = (const float*)d_in[0];
  const long long* ei  = (const long long*)d_in[1];  // int64 per reference
  const float*     W1l = (const float*)d_in[2];
  const float*     b1l = (const float*)d_in[3];
  const float*     W1r = (const float*)d_in[4];
  const float*     W2l = (const float*)d_in[5];
  const float*     b2l = (const float*)d_in[6];
  const float*     W2r = (const float*)d_in[7];
  float* out = (float*)d_out;

  const int       D  = 128;
  const int       N  = in_sizes[0] / D;
  const long long E  = (long long)(in_sizes[1] / 2);
  const int       WN = D * D;  // 16384 weights per matrix

  // Workspace: agg[N*128 f32] | deg[N f32] | xbf[N*128 bf16] | hbf[N*128 bf16] | 4x W bf16
  char* ws = (char*)d_ws;
  const size_t aggBytes = (size_t)N * D * sizeof(float);
  const size_t degBytes = (size_t)N * sizeof(float);
  const size_t fbfBytes = (size_t)N * D * sizeof(unsigned short);
  float* agg = (float*)ws;
  float* deg = (float*)(ws + aggBytes);
  const size_t xOff = (aggBytes + degBytes + 255) & ~(size_t)255;
  unsigned short* xbf = (unsigned short*)(ws + xOff);
  const size_t hOff = (xOff + fbfBytes + 255) & ~(size_t)255;
  unsigned short* hbf = (unsigned short*)(ws + hOff);
  const size_t wOff = (hOff + fbfBytes + 255) & ~(size_t)255;
  unsigned short* w1l = (unsigned short*)(ws + wOff);
  unsigned short* w1r = w1l + WN;
  unsigned short* w2l = w1r + WN;
  unsigned short* w2r = w2l + WN;

  // 0) Pre-convert weights and x to bf16 (one-time; halves per-edge gather
  //    traffic and removes in-GEMM VALU conversion for the root-feature stream)
  cvt_bf16x4<<<(WN / 4 + 255) / 256, 256, 0, stream>>>(W1l, w1l, WN / 4);
  cvt_bf16x4<<<(WN / 4 + 255) / 256, 256, 0, stream>>>(W1r, w1r, WN / 4);
  cvt_bf16x4<<<(WN / 4 + 255) / 256, 256, 0, stream>>>(W2l, w2l, WN / 4);
  cvt_bf16x4<<<(WN / 4 + 255) / 256, 256, 0, stream>>>(W2r, w2r, WN / 4);
  {
    const long long n4 = (long long)N * D / 4;
    cvt_bf16x4<<<(int)((n4 + 255) / 256), 256, 0, stream>>>(x, xbf, n4);
  }

  const int tiles32 = (N + 31) / 32;
  const int gemmBlocks = (tiles32 + 3) / 4;

  // 1) Layer 1: zero agg+deg (contiguous), scatter, fused GEMM -> hbf (relu, bf16)
  {
    const long long zc = (long long)N * D + N;
    zero_f32<<<(int)((zc + 255) / 256), 256, 0, stream>>>(agg, zc);
  }
  sage_scatter_bf<<<(int)((E + 7) / 8), 256, 0, stream>>>(xbf, ei, E, agg, deg, 1);
  sage_gemm<1, 1><<<gemmBlocks, 128, 0, stream>>>(agg, deg, xbf, w1l, w1r, b1l,
                                                  nullptr, hbf, N);

  // 2) Layer 2: re-zero agg only (deg preserved), scatter hbf, fused GEMM -> out (f32)
  {
    const long long zc = (long long)N * D;
    zero_f32<<<(int)((zc + 255) / 256), 256, 0, stream>>>(agg, zc);
  }
  sage_scatter_bf<<<(int)((E + 7) / 8), 256, 0, stream>>>(hbf, ei, E, agg, deg, 0);
  sage_gemm<0, 0><<<gemmBlocks, 128, 0, stream>>>(agg, deg, hbf, w2l, w2r, b2l,
                                                  out, nullptr, N);
}
